// TorchKNN_63720134803806
// MI455X (gfx1250) — matrix operand order repaired and checked by reading, not compiled, not run on hardware
//
#include <hip/hip_runtime.h>

// ---------------------------------------------------------------------------
// Fused KNN (k=7) regression for MI455X (gfx1250, wave32, WMMA, async-LDS).
//   score[q,n] = ||t_n||^2 - 2 * <x_q, t_n>     (x_sq and sqrt are rank-invariant)
//   Kernel 1: fp32 -> bf16 conversion of X / X_train + ||t_n||^2
//   Kernel 2: bf16 WMMA GEMM, double-buffered async B tiles, fused top-7
//   Kernel 3: merge partial top-7s, gather y_train, mean -> predictions
// ---------------------------------------------------------------------------

#define Q_DIM   2048
#define N_DIM   65536
#define D_DIM   768
#define KNN_K   7
#define MTILE   128            // queries per workgroup (8 waves x 16)
#define NSPLIT  32             // N segments (grid.y)
#define SEG     (N_DIM / NSPLIT)   // 2048 train cols per workgroup
#define NSTEP   32             // train cols per n-tile (2 WMMA tiles of 16)
#define KFRAGS  (D_DIM / 32)   // 24 K-fragments of 32
#define KHALF   384            // halfs of K staged per LDS buffer
#define BPITCH  392            // LDS B row pitch in halfs: 384+8 pad -> 196 dw = 4 mod 64
#define SPITCH  17             // LDS score-tile pitch in floats

#if __has_builtin(__builtin_amdgcn_global_load_async_to_lds_b128)
#define HAVE_ASYNC_LDS 1
#endif

typedef __attribute__((ext_vector_type(16))) __bf16        v16bf;
typedef __attribute__((ext_vector_type(8)))  float         v8f;
typedef __attribute__((ext_vector_type(4)))  unsigned int  u32x4;
typedef __attribute__((ext_vector_type(4)))  int           i32x4;

typedef __attribute__((address_space(1))) i32x4 GVec;   // global 16B vector
typedef __attribute__((address_space(3))) i32x4 LVec;   // LDS    16B vector

union FragU { u32x4 q[2]; v16bf v; };   // 32 bytes: two 16B chunks -> one WMMA operand

struct Cand { float s; int i; };        // 8 bytes: score + train index

__device__ __forceinline__ unsigned short f2bf(float f) {
    unsigned int u = __float_as_uint(f);
    unsigned int r = u + 0x7FFFu + ((u >> 16) & 1u);   // round-to-nearest-even
    return (unsigned short)(r >> 16);
}

__device__ __forceinline__ v8f wmma_bf16(v16bf a, v16bf b, v8f c) {
    return __builtin_amdgcn_wmma_f32_16x16x32_bf16(
        /*neg_a=*/false, a, /*neg_b=*/false, b,
        /*c_mod=*/(short)0, c, /*reuse_a=*/false, /*reuse_b=*/false);
}

// Sorted (ascending) insert into a 7-entry register list; fully unrolled selects.
__device__ __forceinline__ void topk_insert(float (&d)[KNN_K], int (&ix)[KNN_K],
                                            float c, int ci) {
    float cs = c; int cc = ci;
    #pragma unroll
    for (int s = 0; s < KNN_K; ++s) {
        bool  sw = cs < d[s];
        float td = d[s]; int ti = ix[s];
        d[s]  = sw ? cs : td;
        ix[s] = sw ? cc : ti;
        cs    = sw ? td : cs;
        cc    = sw ? ti : cc;
    }
}

// ---------------------------------------------------------------------------
// Kernel 1: convert fp32 rows to bf16, compute ||row||^2 for train rows.
// ---------------------------------------------------------------------------
__global__ __launch_bounds__(256)
void knn_prep_kernel(const float* __restrict__ X, const float* __restrict__ Xtr,
                     unsigned short* __restrict__ Xq16,
                     unsigned short* __restrict__ Xtr16,
                     float* __restrict__ tsq) {
    const int row = blockIdx.x;
    const int t   = threadIdx.x;
    const bool train = row < N_DIM;

    const float* src;
    unsigned short* dst;
    if (train) { src = Xtr + (size_t)row * D_DIM;           dst = Xtr16 + (size_t)row * D_DIM; }
    else       { src = X   + (size_t)(row - N_DIM) * D_DIM; dst = Xq16 + (size_t)(row - N_DIM) * D_DIM; }

    float ss = 0.f;
    #pragma unroll
    for (int j = 0; j < D_DIM / 256; ++j) {
        int i = t + j * 256;
        float v = src[i];
        ss = fmaf(v, v, ss);
        dst[i] = f2bf(v);
    }

    __shared__ float red[256];
    red[t] = ss;
    __syncthreads();
    #pragma unroll
    for (int o = 128; o > 0; o >>= 1) {
        if (t < o) red[t] += red[t + o];
        __syncthreads();
    }
    if (train && t == 0) tsq[row] = red[0];
}

// ---------------------------------------------------------------------------
// Kernel 2: fused bf16 WMMA GEMM + per-query top-7 over one N segment.
// Grid: (Q/128, NSPLIT) x 256 threads (8 waves).
//   wave w: 16 resident query rows (A in VGPRs, 24 x v16bf = 192 VGPRs/lane)
//   double-buffered 32-col x 384-half B tiles filled via async-to-LDS DMA.
// ---------------------------------------------------------------------------
__global__ __launch_bounds__(256)
void knn_main_kernel(const unsigned short* __restrict__ Xq16,
                     const unsigned short* __restrict__ Xtr16,
                     const float* __restrict__ tsq,
                     Cand* __restrict__ partials) {
    __shared__ __align__(16) unsigned short Bbuf[2][NSTEP * BPITCH];  // 2 x 25,088 B
    __shared__ float Slds[8 * 16 * SPITCH];                           //     8,704 B

    const int t    = threadIdx.x;
    const int w    = t >> 5;          // wave id 0..7
    const int L    = t & 31;          // lane
    const int qrow = L & 15;          // M (A frag) / N (B frag, C col) index within 16
    const int hv   = L >> 4;          // lane half
    const int qbase   = blockIdx.x * MTILE + w * 16;
    const int segbase = blockIdx.y * SEG;
    const int fr = t >> 3, fc = t & 7;   // B fill: row 0..31, 1/8 of a K-half row

    // --- B-tile fill: 32 rows x 384 halfs (one K-half) into buffer pb -------
#ifdef HAVE_ASYNC_LDS
    #define WAIT_ASYNC() asm volatile("s_wait_asynccnt 0" ::: "memory")
    #define FILL(pb, nn, kh)                                                           \
    {                                                                                  \
        const unsigned short* g = Xtr16 + (size_t)(segbase + (nn) + fr) * D_DIM        \
                                        + (kh) * KHALF + fc * 8;                       \
        unsigned short* l = &Bbuf[pb][fr * BPITCH + fc * 8];                           \
        _Pragma("unroll")                                                              \
        for (int j = 0; j < 6; ++j)                                                    \
            __builtin_amdgcn_global_load_async_to_lds_b128(                            \
                (GVec*)(void*)(g + j * 64),                                            \
                (LVec*)(void*)(l + j * 64), 0, 0);                                     \
    }
#else
    #define WAIT_ASYNC() ((void)0)
    #define FILL(pb, nn, kh)                                                           \
    {                                                                                  \
        const u32x4* src = (const u32x4*)(Xtr16 + (size_t)(segbase + (nn) + fr) * D_DIM \
                                          + (kh) * KHALF);                             \
        _Pragma("unroll")                                                              \
        for (int j = 0; j < 6; ++j) {                                                  \
            int c = fc + 8 * j;                                                        \
            *(u32x4*)(&Bbuf[pb][fr * BPITCH + c * 8]) = src[c];                        \
        }                                                                              \
    }
#endif

    // --- Load A fragments for this wave's 16 queries into registers ---------
    // 16-bit A 16x32 layout: lane m=L%16; elems 0..7 = K 8*hv..+7; 8..15 = K 16+8*hv..+7.
    v16bf afrag[KFRAGS];
    {
        const unsigned short* arow = Xq16 + (size_t)(qbase + qrow) * D_DIM;
        #pragma unroll
        for (int kf = 0; kf < KFRAGS; ++kf) {
            FragU fu;
            fu.q[0] = *(const u32x4*)(arow + kf * 32 + hv * 8);
            fu.q[1] = *(const u32x4*)(arow + kf * 32 + hv * 8 + 16);
            afrag[kf] = fu.v;
        }
    }

    // Running top-7: lane owns (query = L%16, column-half hv).
    float dist[KNN_K]; int ind[KNN_K];
    #pragma unroll
    for (int j = 0; j < KNN_K; ++j) { dist[j] = 3.0e38f; ind[j] = 0; }

    float* sl = &Slds[w * 16 * SPITCH];

    int p = 0;
    FILL(0, 0, 0);   // prime the pipeline

    for (int n0 = 0; n0 < SEG; n0 += NSTEP) {
        v8f acc0 = {0.f,0.f,0.f,0.f,0.f,0.f,0.f,0.f};
        v8f acc1 = {0.f,0.f,0.f,0.f,0.f,0.f,0.f,0.f};

        #pragma unroll
        for (int kh = 0; kh < 2; ++kh) {
            WAIT_ASYNC();        // my previous fill (buffer p) has landed in LDS
            __syncthreads();     // everyone's fill landed; everyone done reading buf p^1

            // issue next stage's fill into the other buffer (overlaps compute)
            {
                int nn = n0, nk = kh + 1;
                if (nk == 2) { nk = 0; nn += NSTEP; }
                if (nn < SEG) FILL(p ^ 1, nn, nk);
            }

            // --- 24 WMMAs on buffer p (12 K-frags x 2 column tiles) ---------
            const unsigned short* bb = &Bbuf[p][0];
            #pragma unroll
            for (int kfl = 0; kfl < 12; ++kfl) {
                const int kf = kh * 12 + kfl;
                // 16-bit B 32x16 layout: lane n=L%16; elems 0..7 = K 8*hv..+7; 8..15 = K 16+8*hv..+7.
                const int kk = kfl * 32 + hv * 8;
                FragU b0, b1;
                const unsigned short* bp0 = bb + qrow * BPITCH + kk;
                const unsigned short* bp1 = bb + (16 + qrow) * BPITCH + kk;
                b0.q[0] = *(const u32x4*)bp0; b0.q[1] = *(const u32x4*)(bp0 + 16);
                b1.q[0] = *(const u32x4*)bp1; b1.q[1] = *(const u32x4*)(bp1 + 16);
                acc0 = wmma_bf16(afrag[kf], b0.v, acc0);
                acc1 = wmma_bf16(afrag[kf], b1.v, acc1);
            }
            p ^= 1;
        }

        // --- selection: stage scores through LDS, threshold-guarded insert --
        // C layout: lane col n = L%16, row m = r + 8*hv.
        {
            float ts = tsq[segbase + n0 + qrow];          // tile-0 column norm
            #pragma unroll
            for (int r = 0; r < 8; ++r)
                sl[(r + 8 * hv) * SPITCH + qrow] = fmaf(-2.f, acc0[r], ts);
            __builtin_amdgcn_wave_barrier();
            #pragma unroll
            for (int j = 0; j < 8; ++j) {
                int col = 8 * hv + j;
                float c = sl[qrow * SPITCH + col];
                if (c < dist[KNN_K - 1]) topk_insert(dist, ind, c, segbase + n0 + col);
            }
            __builtin_amdgcn_wave_barrier();

            float ts1 = tsq[segbase + n0 + 16 + qrow];    // tile-1 column norm
            #pragma unroll
            for (int r = 0; r < 8; ++r)
                sl[(r + 8 * hv) * SPITCH + qrow] = fmaf(-2.f, acc1[r], ts1);
            __builtin_amdgcn_wave_barrier();
            #pragma unroll
            for (int j = 0; j < 8; ++j) {
                int col = 8 * hv + j;
                float c = sl[qrow * SPITCH + col];
                if (c < dist[KNN_K - 1]) topk_insert(dist, ind, c, segbase + n0 + 16 + col);
            }
            __builtin_amdgcn_wave_barrier();
        }
    }

    // --- emit this (query, half)'s partial top-7 ----------------------------
    Cand* out = partials + ((size_t)(qbase + qrow) * NSPLIT + blockIdx.y) * (2 * KNN_K) + hv * KNN_K;
    #pragma unroll
    for (int j = 0; j < KNN_K; ++j) { out[j].s = dist[j]; out[j].i = ind[j]; }
}

// ---------------------------------------------------------------------------
// Kernel 3: per query, merge NSPLIT*14 candidates -> final 7, gather y, mean.
// ---------------------------------------------------------------------------
__global__ __launch_bounds__(32)
void knn_reduce_kernel(const Cand* __restrict__ partials,
                       const float* __restrict__ y_train,
                       float* __restrict__ out) {
    __shared__ Cand all[32 * KNN_K];
    const int q = blockIdx.x;
    const int L = threadIdx.x;
    const int ncand = NSPLIT * 2 * KNN_K;   // 448

    const Cand* p = partials + (size_t)q * ncand;
    float d[KNN_K]; int ix[KNN_K];
    #pragma unroll
    for (int j = 0; j < KNN_K; ++j) { d[j] = 3.0e38f; ix[j] = 0; }

    for (int j = L; j < ncand; j += 32) {
        Cand c = p[j];
        if (c.s < d[KNN_K - 1]) topk_insert(d, ix, c.s, c.i);
    }
    #pragma unroll
    for (int j = 0; j < KNN_K; ++j) { all[L * KNN_K + j].s = d[j]; all[L * KNN_K + j].i = ix[j]; }
    __syncthreads();

    if (L == 0) {
        float fd[KNN_K]; int fi[KNN_K];
        #pragma unroll
        for (int j = 0; j < KNN_K; ++j) { fd[j] = 3.0e38f; fi[j] = 0; }
        for (int j = 0; j < 32 * KNN_K; ++j) {
            Cand c = all[j];
            if (c.s < fd[KNN_K - 1]) topk_insert(fd, fi, c.s, c.i);
        }
        float s = 0.f;
        #pragma unroll
        for (int j = 0; j < KNN_K; ++j) s += y_train[fi[j]];
        out[q] = s * (1.0f / (float)KNN_K);
    }
}

// ---------------------------------------------------------------------------
extern "C" void kernel_launch(void* const* d_in, const int* in_sizes, int n_in,
                              void* d_out, int out_size, void* d_ws, size_t ws_size,
                              hipStream_t stream) {
    (void)in_sizes; (void)n_in; (void)out_size; (void)ws_size;
    const float* X  = (const float*)d_in[0];   // [Q, D]
    const float* Xt = (const float*)d_in[1];   // [N, D]
    const float* y  = (const float*)d_in[2];   // [N, 1]
    float* out = (float*)d_out;                // [Q, 1]

    char* ws = (char*)d_ws;
    unsigned short* Xtr16 = (unsigned short*)(ws);                                   // 100,663,296 B
    unsigned short* Xq16  = (unsigned short*)(ws + (size_t)N_DIM * D_DIM * 2);       //   3,145,728 B
    float*          tsq   = (float*)(ws + (size_t)(N_DIM + Q_DIM) * D_DIM * 2);      //     262,144 B
    Cand*           parts = (Cand*)(ws + (size_t)(N_DIM + Q_DIM) * D_DIM * 2
                                       + (size_t)N_DIM * 4);                         //   7,340,032 B

    knn_prep_kernel<<<dim3(N_DIM + Q_DIM), dim3(256), 0, stream>>>(X, Xt, Xq16, Xtr16, tsq);
    knn_main_kernel<<<dim3(Q_DIM / MTILE, NSPLIT), dim3(256), 0, stream>>>(Xq16, Xtr16, tsq, parts);
    knn_reduce_kernel<<<dim3(Q_DIM), dim3(32), 0, stream>>>(parts, y, out);
}
